// MoPEBlock_33148557590992
// MI455X (gfx1250) — matrix operand order repaired and checked by reading, not compiled, not run on hardware
//
#include <hip/hip_runtime.h>
#include <hip/hip_bf16.h>
#include <cstdint>

// ---------------- problem constants ----------------
constexpr int NTOK = 4 * 4096;   // B*T = 16384 tokens
constexpr int C    = 1024;
constexpr int H    = 4096;       // 4*C
constexpr int NE   = 8;
constexpr int TM   = 32;               // tokens per FFN tile
constexpr int TILES_PER_E = NTOK / TM; // 512

// LDS row strides (in u16 elements); 1032*2=2064B and 136*2=272B are 16B-aligned
constexpr int XS = 1032;   // rows of 1024 bf16 (+8 pad)
constexpr int SS = 136;    // rows of 128 bf16  (+8 pad)

typedef unsigned int  u32;
typedef unsigned short u16;
typedef unsigned long long u64;
typedef __bf16 bf16_t;
typedef __attribute__((ext_vector_type(16))) bf16_t v16bf;
typedef __attribute__((ext_vector_type(8)))  float  v8f;

union BFrag { v16bf v; u32 u[8]; uint4 q[2]; };

// workspace byte offsets for bf16 mirrors
constexpr size_t XBF_OFF = 2u << 20;                          // 2 MB (lists/probs below)
constexpr size_t W1_OFF  = XBF_OFF + (size_t)NTOK * C * 2;    // + 32 MiB
constexpr size_t W2_OFF  = W1_OFF + (size_t)NE * C * H * 2;   // + 64 MiB

// ---------------- helpers ----------------
__device__ __forceinline__ u16 f2bf(float f) {
  u32 u = __builtin_bit_cast(u32, f);
  u32 r = u + 0x7FFFu + ((u >> 16) & 1u);   // round-to-nearest-even
  return (u16)(r >> 16);
}
__device__ __forceinline__ u32 pk2(float a, float b) {
  return (u32)f2bf(a) | ((u32)f2bf(b) << 16);
}
__device__ __forceinline__ float gelu_exact(float v) {
  return 0.5f * v * (1.0f + erff(v * 0.70710678118654752f));
}

// CDNA5 async global->LDS copy (16B per lane), tracked by ASYNCcnt.
// GV mode: vdst = LDS byte address, vaddr = 64-bit global address, saddr = off.
__device__ __forceinline__ void async_cp16(const void* g, void* l) {
  asm volatile("global_load_async_to_lds_b128 %0, %1, off"
               :: "v"((u32)(uintptr_t)l), "v"((u64)(uintptr_t)g)
               : "memory");
}
__device__ __forceinline__ void async_wait0() {
#if __has_builtin(__builtin_amdgcn_s_wait_asynccnt)
  __builtin_amdgcn_s_wait_asynccnt(0);
#else
  asm volatile("s_wait_asynccnt 0" ::: "memory");
#endif
}

// A-matrix fragment (16x32 bf16) per ISA 7.12.2: two 16B halves per lane
__device__ __forceinline__ void load_a_frag(const u16* base, int stride,
                                            int mrow0, int kb, int lane, BFrag& f) {
  int half = lane >> 4;
  int r = mrow0 + (lane & 15);
  const uint4* q = (const uint4*)(base + (size_t)r * stride + kb + half * 8);
  f.q[0] = q[0];        // K = base .. base+7   (packed pairs)
  f.q[1] = q[2];        // +32B -> K+16
}

// B-matrix fragment (32x16 bf16): lane = N column (lane%16); K = 8h+s (s<8), 16+8h+(s-8) (s>=8)
__device__ __forceinline__ void load_b_frag(const u16* base, int stride,
                                            int ncol0, int lane, BFrag& f) {
  const u16* p = base + ncol0 + (lane & 15);
  int h8 = (lane >> 4) * 8;
#pragma unroll
  for (int i = 0; i < 8; ++i) {
    int k = (i < 4) ? (h8 + 2 * i) : (16 + h8 + 2 * (i - 4));
    f.u[i] = (u32)p[(size_t)k * stride] | ((u32)p[(size_t)(k + 1) * stride] << 16);
  }
}

__device__ __forceinline__ v8f wmma_bf16(const BFrag& a, const BFrag& b, v8f c) {
  return __builtin_amdgcn_wmma_f32_16x16x32_bf16(false, a.v, false, b.v,
                                                 (short)0, c, false, false);
}

// ---------------- kernel 0: fp32 -> bf16 bulk convert (pairs) -------------
__global__ void cvt_bf16_kernel(const float* __restrict__ src,
                                u32* __restrict__ dst, size_t npairs) {
  size_t i = (size_t)blockIdx.x * blockDim.x + threadIdx.x;
  size_t stride = (size_t)gridDim.x * blockDim.x;
  for (; i < npairs; i += stride) {
    float2 v = *(const float2*)(src + 2 * i);
    dst[i] = pk2(v.x, v.y);
  }
}

// ---------------- kernel 1: zero output + counters ----------------
__global__ void zero_kernel(float* __restrict__ out, size_t n, int* __restrict__ counts) {
  size_t i = (size_t)blockIdx.x * blockDim.x + threadIdx.x;
  size_t stride = (size_t)gridDim.x * blockDim.x;
  for (; i < n; i += stride) out[i] = 0.0f;
  if (blockIdx.x == 0 && threadIdx.x < NE) counts[threadIdx.x] = 0;
}

// ---------------- kernel 2: router (one wave32 per token) ----------------
__global__ __launch_bounds__(256) void router_kernel(
    const float* __restrict__ x, const float* __restrict__ mass,
    const float* __restrict__ gw, const float* __restrict__ mb,
    float* __restrict__ probs, int* __restrict__ counts,
    int* __restrict__ tok_list, float* __restrict__ coef_list) {
  const int lane = threadIdx.x & 31;
  const int wid  = threadIdx.x >> 5;
  const int t    = blockIdx.x * 8 + wid;   // NTOK multiple of 8

  float acc[NE];
#pragma unroll
  for (int e = 0; e < NE; ++e) acc[e] = 0.0f;

  const float* xrow = x + (size_t)t * C;
  for (int c = lane; c < C; c += 32) {
    float xv = xrow[c];
#pragma unroll
    for (int e = 0; e < NE; ++e) acc[e] = fmaf(xv, gw[e * C + c], acc[e]);
  }
#pragma unroll
  for (int e = 0; e < NE; ++e) {
#pragma unroll
    for (int off = 16; off > 0; off >>= 1) acc[e] += __shfl_xor(acc[e], off, 32);
  }

  float m = mass[t];
  float p[NE];
  float mx = -3.4e38f;
#pragma unroll
  for (int e = 0; e < NE; ++e) { p[e] = acc[e] + m * mb[e]; mx = fmaxf(mx, p[e]); }
  float s = 0.0f;
#pragma unroll
  for (int e = 0; e < NE; ++e) { p[e] = expf(p[e] - mx); s += p[e]; }
  float inv = 1.0f / s;
#pragma unroll
  for (int e = 0; e < NE; ++e) p[e] *= inv;

  int i0 = 0; float w0 = p[0];
#pragma unroll
  for (int e = 1; e < NE; ++e) if (p[e] > w0) { w0 = p[e]; i0 = e; }
  int i1 = (i0 == 0) ? 1 : 0; float w1 = p[i1];
#pragma unroll
  for (int e = 0; e < NE; ++e) if (e != i0 && p[e] > w1) { w1 = p[e]; i1 = e; }

  if (lane == 0) {
#pragma unroll
    for (int e = 0; e < NE; ++e) probs[(size_t)t * NE + e] = p[e];
    int pos = atomicAdd(&counts[i0], 1);
    tok_list[i0 * NTOK + pos] = t; coef_list[i0 * NTOK + pos] = w0;
    pos = atomicAdd(&counts[i1], 1);
    tok_list[i1 * NTOK + pos] = t; coef_list[i1 * NTOK + pos] = w1;
  }
}

// ---------------- kernel 3: aux loss (deterministic tree reduction) --------
__global__ __launch_bounds__(256) void aux_kernel(const float* __restrict__ probs,
                                                  float* __restrict__ out_aux) {
  __shared__ float red[256];
  const int t = threadIdx.x;
  float cs[NE];
#pragma unroll
  for (int e = 0; e < NE; ++e) cs[e] = 0.0f;
  for (int i = t; i < NTOK; i += 256) {
#pragma unroll
    for (int e = 0; e < NE; ++e) cs[e] += probs[(size_t)i * NE + e];
  }
  __shared__ float tot[NE];
  for (int e = 0; e < NE; ++e) {
    red[t] = cs[e];
    __syncthreads();
    for (int off = 128; off > 0; off >>= 1) {
      if (t < off) red[t] += red[t + off];
      __syncthreads();
    }
    if (t == 0) tot[e] = red[0];
    __syncthreads();
  }
  if (t == 0) {
    const float target = (float)NTOK / (float)NE;   // 2048
    float a = 0.0f;
#pragma unroll
    for (int e = 0; e < NE; ++e) { float d = tot[e] - target; a += d * d; }
    out_aux[0] = a / (float)NE;
  }
}

// ---------------- kernel 4: fused expert FFN with WMMA (bf16 inputs) -------
__global__ __launch_bounds__(256) void ffn_kernel(
    const u16* __restrict__ xbf,
    const u16* __restrict__ w1bf, const float* __restrict__ b1,
    const u16* __restrict__ w2bf, const float* __restrict__ b2,
    const int* __restrict__ counts, const int* __restrict__ tok_list,
    const float* __restrict__ coef_list, float* __restrict__ out) {
  const int e    = blockIdx.x >> 9;          // TILES_PER_E == 512
  const int tile = blockIdx.x & (TILES_PER_E - 1);
  const int cnt  = counts[e];
  if (tile * TM >= cnt) return;              // block-uniform exit

  extern __shared__ unsigned char smem[];
  u16*   xs    = (u16*)smem;                 // [32][XS]  x tile
  u16*   w2s   = xs  + 32 * XS;              // [32][XS]  W2 k-slice
  u16*   w1s   = w2s + 32 * XS;              // [32][SS]  W1 k-slice
  u16*   acts  = w1s + 32 * SS;              // [32][SS]  gelu act chunk
  int*   toks  = (int*)(acts + 32 * SS);     // [32]
  float* coefs = (float*)(toks + TM);        // [32]

  const int tid  = threadIdx.x;
  const int lane = tid & 31;
  const int wid  = tid >> 5;
  const int mt   = wid >> 2;                 // 0..1   (M tile of 16)
  const int cB   = wid & 3;                  // 0..3   (N column group)

  if (tid < TM) {
    int slot = tile * TM + tid;
    if (slot < cnt) {
      toks[tid]  = tok_list[e * NTOK + slot];
      coefs[tid] = coef_list[e * NTOK + slot];
    } else { toks[tid] = 0; coefs[tid] = 0.0f; }
  }
  __syncthreads();

  // Gather X tile [32][1024] bf16 -> LDS via async global->LDS copies
  {
    int r = tid >> 3;
    int c0 = (tid & 7) * 128;
    const u16* src = xbf + (size_t)toks[r] * C + c0;
    u16* dst = xs + (size_t)r * XS + c0;
#pragma unroll
    for (int i = 0; i < 16; ++i) async_cp16(src + i * 8, dst + i * 8);
  }

  const u16*   W1e = w1bf + (size_t)e * C * H;
  const u16*   W2e = w2bf + (size_t)e * H * C;
  const float* b1e = b1 + (size_t)e * H;
  const float* b2e = b2 + (size_t)e * C;

  // Register-resident output accumulators: wave owns (mt, nt = cB+4j), j=0..15
  v8f accB[16];
#pragma unroll
  for (int j = 0; j < 16; ++j)
    accB[j] = (v8f){0.f, 0.f, 0.f, 0.f, 0.f, 0.f, 0.f, 0.f};

  for (int hc = 0; hc < H; hc += 128) {
    // ---------------- Phase A: act = gelu(X @ W1[:, hc:hc+128] + b1) -------
    v8f accA0 = (v8f){0.f, 0.f, 0.f, 0.f, 0.f, 0.f, 0.f, 0.f};
    v8f accA1 = accA0;
    for (int kb = 0; kb < C; kb += 32) {
      __syncthreads();
      { // stage W1 slice [kb..kb+31][hc..hc+127] via async copy
        int r = tid >> 3, c0 = (tid & 7) * 16;
        const u16* src = W1e + (size_t)(kb + r) * H + hc + c0;
        u16* dst = w1s + (size_t)r * SS + c0;
        async_cp16(src, dst);
        async_cp16(src + 8, dst + 8);
      }
      async_wait0();
      __syncthreads();
      BFrag a; load_a_frag(xs, XS, mt * 16, kb, lane, a);
      BFrag b0, b1f;
      load_b_frag(w1s, SS, cB * 16, lane, b0);
      load_b_frag(w1s, SS, (cB + 4) * 16, lane, b1f);
      accA0 = wmma_bf16(a, b0, accA0);
      accA1 = wmma_bf16(a, b1f, accA1);
    }
    // epilogue: +b1, gelu, store bf16 act tile
    {
      int colb  = lane & 15;
      int mbase = mt * 16 + (lane >> 4) * 8;
      int n0 = cB * 16 + colb;
      int n1 = (cB + 4) * 16 + colb;
      float b1v0 = b1e[hc + n0];
      float b1v1 = b1e[hc + n1];
#pragma unroll
      for (int r = 0; r < 8; ++r) {
        acts[(size_t)(mbase + r) * SS + n0] = f2bf(gelu_exact(accA0[r] + b1v0));
        acts[(size_t)(mbase + r) * SS + n1] = f2bf(gelu_exact(accA1[r] + b1v1));
      }
    }
    // ---------------- Phase B: out += act @ W2[hc:hc+128, :] ---------------
    for (int kk = 0; kk < 128; kk += 32) {
      __syncthreads();   // act visible; w2s reuse safe
      { // stage W2 slice [hc+kk .. +31][0..1023] via async copy
        int r = tid >> 3, c0 = (tid & 7) * 128;
        const u16* src = W2e + (size_t)(hc + kk + r) * C + c0;
        u16* dst = w2s + (size_t)r * XS + c0;
#pragma unroll
        for (int i = 0; i < 16; ++i) async_cp16(src + i * 8, dst + i * 8);
      }
      async_wait0();
      __syncthreads();
      BFrag a; load_a_frag(acts, SS, mt * 16, kk, lane, a);
      // software-pipelined B fragments: load j+1 while computing j
      BFrag bf[2];
      load_b_frag(w2s, XS, cB * 16, lane, bf[0]);
#pragma unroll
      for (int j = 0; j < 16; ++j) {
        if (j < 15)
          load_b_frag(w2s, XS, (cB + 4 * (j + 1)) * 16, lane, bf[(j + 1) & 1]);
        accB[j] = wmma_bf16(a, bf[j & 1], accB[j]);
      }
    }
  }

  // ---------------- writeout: +b2, scale by routing weight, atomic combine --
  {
    int colb  = lane & 15;
    int mbase = mt * 16 + (lane >> 4) * 8;
#pragma unroll
    for (int j = 0; j < 16; ++j) {
      int col = (cB + 4 * j) * 16 + colb;
      float b2v = b2e[col];
#pragma unroll
      for (int r = 0; r < 8; ++r) {
        int mrow = mbase + r;
        float contrib = coefs[mrow] * (accB[j][r] + b2v);
        atomicAdd(out + (size_t)toks[mrow] * C + col, contrib);
      }
    }
  }
}

// ---------------- launcher ----------------
extern "C" void kernel_launch(void* const* d_in, const int* in_sizes, int n_in,
                              void* d_out, int out_size, void* d_ws, size_t ws_size,
                              hipStream_t stream) {
  (void)in_sizes; (void)n_in; (void)out_size; (void)ws_size;
  const float* x    = (const float*)d_in[0];   // [N, C]
  const float* mass = (const float*)d_in[1];   // [N, 1]
  const float* gw   = (const float*)d_in[2];   // [E, C]
  const float* mb   = (const float*)d_in[3];   // [E]
  const float* W1   = (const float*)d_in[4];   // [E, C, H]
  const float* b1   = (const float*)d_in[5];   // [E, H]
  const float* W2   = (const float*)d_in[6];   // [E, H, C]
  const float* b2   = (const float*)d_in[7];   // [E, C]
  float* out = (float*)d_out;                  // [N*C] results + [1] aux

  unsigned char* ws = (unsigned char*)d_ws;
  // small workspace region (< 2 MB)
  int*   counts = (int*)ws;                                   // [8] (+pad)
  int*   toks   = (int*)ws + 16;                              // [E][N]
  float* coefs  = (float*)((int*)ws + 16 + NE * NTOK);        // [E][N]
  float* probs  = coefs + (size_t)NE * NTOK;                  // [N][E]
  // bf16 mirrors
  u16* xbf  = (u16*)(ws + XBF_OFF);
  u16* w1bf = (u16*)(ws + W1_OFF);
  u16* w2bf = (u16*)(ws + W2_OFF);

  const size_t out_elems = (size_t)NTOK * C + 1;

  // bulk fp32 -> bf16 conversion (once per launch, memory bound)
  cvt_bf16_kernel<<<4096, 256, 0, stream>>>(x,  (u32*)xbf,  (size_t)NTOK * C / 2);
  cvt_bf16_kernel<<<4096, 256, 0, stream>>>(W1, (u32*)w1bf, (size_t)NE * C * H / 2);
  cvt_bf16_kernel<<<4096, 256, 0, stream>>>(W2, (u32*)w2bf, (size_t)NE * H * C / 2);

  zero_kernel<<<2048, 256, 0, stream>>>(out, out_elems, counts);
  router_kernel<<<NTOK / 8, 256, 0, stream>>>(x, mass, gw, mb,
                                              probs, counts, toks, coefs);
  aux_kernel<<<1, 256, 0, stream>>>(probs, out + (size_t)NTOK * C);

  const size_t smem = (size_t)(32 * XS) * 2 * 2   // xs + w2s
                    + (size_t)(32 * SS) * 2 * 2   // w1s + acts
                    + TM * sizeof(int) + TM * sizeof(float);
  ffn_kernel<<<NE * TILES_PER_E, 256, smem, stream>>>(xbf, w1bf, b1, w2bf, b2,
                                                      counts, toks, coefs, out);
}
